// GIN_38560216383777
// MI455X (gfx1250) — matrix operand order repaired and checked by reading, not compiled, not run on hardware
//
#include <hip/hip_runtime.h>
#include <hip/hip_bf16.h>

typedef __attribute__((ext_vector_type(16))) __bf16 v16bf;
typedef __attribute__((ext_vector_type(8)))  float  v8f;
typedef __attribute__((ext_vector_type(4)))  float  v4f;

#define N_NODES  20000
#define N_EDGES  640000
#define F_IN     128
#define HID      256
#define NCLS     40
#define NCLS_PAD 48
#define MTILES   (N_NODES / 16)   // 1250, exact

// ---------------------------------------------------------------- utilities
__global__ void zero_f32(float* __restrict__ p, int n) {
    int t = blockIdx.x * blockDim.x + threadIdx.x;
    if (t < n) p[t] = 0.f;
}

// Pack W (K x Ncols, f32 row-major) into the WMMA B-operand register order,
// bf16, with columns zero-padded to NcolsPad (multiple of 16).
// Layout: Bp[((nt*kblocks + kb)*32 + lane)*16 + i] = W[kb*32 + 8*(lane>=16) + (i&7) + 16*(i>=8)][nt*16 + (lane&15)]
__global__ void pack_w_bf16(const float* __restrict__ W, __bf16* __restrict__ Bp,
                            int K, int Ncols, int NcolsPad) {
    int t = blockIdx.x * blockDim.x + threadIdx.x;
    int kblocks = K >> 5;
    int total = (NcolsPad >> 4) * kblocks * 32 * 16;
    if (t >= total) return;
    int i    = t & 15;
    int lane = (t >> 4) & 31;
    int blk  = t >> 9;              // nt*kblocks + kb
    int kb   = blk % kblocks;
    int nt   = blk / kblocks;
    int n = nt * 16 + (lane & 15);
    int k = kb * 32 + ((lane >> 4) ? 8 : 0) + (i & 7) + ((i & 8) ? 16 : 0);
    float v = (n < Ncols) ? W[(size_t)k * Ncols + n] : 0.f;
    Bp[t] = (__bf16)v;
}

// agg[dst[e], :] += feat[src[e], :]   (float4 chunks, hardware fp32 atomics)
__global__ void scatter_add_f32(const float* __restrict__ feat,
                                const long long* __restrict__ ei,   // [2, N_EDGES] int64
                                float* __restrict__ agg, int D4, int D) {
    long long t = (long long)blockIdx.x * blockDim.x + threadIdx.x;
    long long total = (long long)N_EDGES * D4;
    if (t >= total) return;
    int e = (int)(t / D4);
    int c = (int)(t % D4);
    int s = (int)ei[e];
    int d = (int)ei[N_EDGES + e];
    v4f v = *(const v4f*)(feat + (size_t)s * D + c * 4);
    float* dp = agg + (size_t)d * D + c * 4;
    __hip_atomic_fetch_add(dp + 0, v[0], __ATOMIC_RELAXED, __HIP_MEMORY_SCOPE_AGENT);
    __hip_atomic_fetch_add(dp + 1, v[1], __ATOMIC_RELAXED, __HIP_MEMORY_SCOPE_AGENT);
    __hip_atomic_fetch_add(dp + 2, v[2], __ATOMIC_RELAXED, __HIP_MEMORY_SCOPE_AGENT);
    __hip_atomic_fetch_add(dp + 3, v[3], __ATOMIC_RELAXED, __HIP_MEMORY_SCOPE_AGENT);
}

// OUT[m, n] = relu?( (X[m,:] (+AGG[m,:])) @ Wpack + bias[n] )
// One wave per 16x16 output tile; v_wmma_f32_16x16x32_bf16 over K.
// HAS_AGG is a compile-time flag so the K-loop has no EXEC-mask manipulation.
template <bool HAS_AGG>
__global__ void gin_gemm_bf16(const float* __restrict__ X,
                              const float* __restrict__ AGG,      // used iff HAS_AGG
                              const __bf16* __restrict__ Bp,      // packed weights
                              const float* __restrict__ bias,     // nullable
                              float* __restrict__ OUT,
                              int Kdim, int NcolsPad, int NcolsOut, int relu) {
    const int ntiles = NcolsPad >> 4;
    const int wave = blockIdx.x * (blockDim.x >> 5) + (threadIdx.x >> 5);
    if (wave >= MTILES * ntiles) return;             // wave-uniform: EXEC stays all-ones
    const int mt = wave / ntiles;
    const int nt = wave - mt * ntiles;
    const int lane = threadIdx.x & 31;
    const int half = lane >> 4;
    const int r = lane & 15;
    const int kblocks = Kdim >> 5;

    const float* xrow = X + (size_t)(mt * 16 + r) * Kdim;
    const float* arow = HAS_AGG ? AGG + (size_t)(mt * 16 + r) * Kdim : nullptr;
    const __bf16* bpp = Bp + (((size_t)nt * kblocks * 32 + lane) << 4);

    v8f acc = {};
    for (int kb = 0; kb < kblocks; ++kb) {
        const int klo = kb * 32 + half * 8;
        v4f a0 = *(const v4f*)(xrow + klo);
        v4f a1 = *(const v4f*)(xrow + klo + 4);
        v4f a2 = *(const v4f*)(xrow + klo + 16);
        v4f a3 = *(const v4f*)(xrow + klo + 20);
        if constexpr (HAS_AGG) {
            a0 += *(const v4f*)(arow + klo);
            a1 += *(const v4f*)(arow + klo + 4);
            a2 += *(const v4f*)(arow + klo + 16);
            a3 += *(const v4f*)(arow + klo + 20);
        }
        v16bf A;
        A[0]  = (__bf16)a0[0]; A[1]  = (__bf16)a0[1]; A[2]  = (__bf16)a0[2]; A[3]  = (__bf16)a0[3];
        A[4]  = (__bf16)a1[0]; A[5]  = (__bf16)a1[1]; A[6]  = (__bf16)a1[2]; A[7]  = (__bf16)a1[3];
        A[8]  = (__bf16)a2[0]; A[9]  = (__bf16)a2[1]; A[10] = (__bf16)a2[2]; A[11] = (__bf16)a2[3];
        A[12] = (__bf16)a3[0]; A[13] = (__bf16)a3[1]; A[14] = (__bf16)a3[2]; A[15] = (__bf16)a3[3];

        v16bf B = *(const v16bf*)(bpp + ((size_t)kb << 9));   // kb*32*16 bf16

        acc = __builtin_amdgcn_wmma_f32_16x16x32_bf16(
            /*neg_a=*/false, A, /*neg_b=*/false, B,
            /*c_mod=*/(short)0, acc, /*reuse_a=*/false, /*reuse_b=*/false);
    }

    const int ncol = nt * 16 + r;
    if (ncol < NcolsOut) {                            // per-lane guard AFTER all WMMA
        const float bv = bias ? bias[ncol] : 0.f;
        #pragma unroll
        for (int i = 0; i < 8; ++i) {
            const int row = mt * 16 + i + half * 8;
            float v = acc[i] + bv;
            if (relu) v = v > 0.f ? v : 0.f;
            OUT[(size_t)row * NcolsOut + ncol] = v;
        }
    }
}

// Final layer-4 combine: out = y + agg(y) + b4   (gemm-first restructuring)
__global__ void combine_out(const float* __restrict__ y, const float* __restrict__ agg,
                            const float* __restrict__ bias, float* __restrict__ out) {
    int t = blockIdx.x * blockDim.x + threadIdx.x;
    if (t >= N_NODES * NCLS) return;
    int j = t % NCLS;
    out[t] = y[t] + agg[t] + bias[j];
}

// ---------------------------------------------------------------- launch
extern "C" void kernel_launch(void* const* d_in, const int* in_sizes, int n_in,
                              void* d_out, int out_size, void* d_ws, size_t ws_size,
                              hipStream_t stream) {
    (void)in_sizes; (void)n_in; (void)out_size; (void)ws_size;
    const float*     x  = (const float*)d_in[0];
    const long long* ei = (const long long*)d_in[1];      // int64 edge_index [2, E]
    const float* W1 = (const float*)d_in[2]; const float* b1 = (const float*)d_in[3];
    const float* W2 = (const float*)d_in[4]; const float* b2 = (const float*)d_in[5];
    const float* W3 = (const float*)d_in[6]; const float* b3 = (const float*)d_in[7];
    const float* W4 = (const float*)d_in[8]; const float* b4 = (const float*)d_in[9];
    float* out = (float*)d_out;

    // workspace carve-up
    char* ws = (char*)d_ws;
    size_t off = 0;
    auto take = [&](size_t bytes) { char* p = ws + off; off = (off + bytes + 255) & ~(size_t)255; return p; };
    float*  hA   = (float*)take((size_t)N_NODES * HID * 4);
    float*  hB   = (float*)take((size_t)N_NODES * HID * 4);
    float*  agg  = (float*)take((size_t)N_NODES * HID * 4);
    float*  y4   = (float*)take((size_t)N_NODES * NCLS * 4);
    float*  agg4 = (float*)take((size_t)N_NODES * NCLS * 4);
    __bf16* W1p  = (__bf16*)take((size_t)F_IN * HID * 2);
    __bf16* W2p  = (__bf16*)take((size_t)HID * HID * 2);
    __bf16* W3p  = (__bf16*)take((size_t)HID * HID * 2);
    __bf16* W4p  = (__bf16*)take((size_t)HID * NCLS_PAD * 2);

    const int TB = 256;
    const int WAVES_PER_BLK = TB / 32;

    // pack weights -> bf16 B-operand order (tiny, once per launch)
    {
        int t1 = (HID / 16) * (F_IN / 32) * 512;
        pack_w_bf16<<<(t1 + TB - 1) / TB, TB, 0, stream>>>(W1, W1p, F_IN, HID, HID);
        int t2 = (HID / 16) * (HID / 32) * 512;
        pack_w_bf16<<<(t2 + TB - 1) / TB, TB, 0, stream>>>(W2, W2p, HID, HID, HID);
        pack_w_bf16<<<(t2 + TB - 1) / TB, TB, 0, stream>>>(W3, W3p, HID, HID, HID);
        int t4 = (NCLS_PAD / 16) * (HID / 32) * 512;
        pack_w_bf16<<<(t4 + TB - 1) / TB, TB, 0, stream>>>(W4, W4p, HID, NCLS, NCLS_PAD);
    }

    auto launch_layer = [&](const float* in, int Din, const __bf16* Wp, const float* bias,
                            float* h_out) {
        // zero + scatter (Din wide) + fused (in+agg)@W+b, relu
        int nz = N_NODES * Din;
        zero_f32<<<(nz + TB - 1) / TB, TB, 0, stream>>>(agg, nz);
        long long st = (long long)N_EDGES * (Din / 4);
        scatter_add_f32<<<(unsigned)((st + TB - 1) / TB), TB, 0, stream>>>(in, ei, agg, Din / 4, Din);
        int waves = MTILES * (HID / 16);
        gin_gemm_bf16<true><<<(waves + WAVES_PER_BLK - 1) / WAVES_PER_BLK, TB, 0, stream>>>(
            in, agg, Wp, bias, h_out, Din, HID, HID, /*relu=*/1);
    };

    launch_layer(x,  F_IN, W1p, b1, hA);   // layer 1 (aggregate at width 128)
    launch_layer(hA, HID,  W2p, b2, hB);   // layer 2
    launch_layer(hB, HID,  W3p, b3, hA);   // layer 3

    // layer 4: transform first (256 -> 40), then aggregate only 40-wide
    {
        int waves = MTILES * (NCLS_PAD / 16);
        gin_gemm_bf16<false><<<(waves + WAVES_PER_BLK - 1) / WAVES_PER_BLK, TB, 0, stream>>>(
            hA, nullptr, W4p, nullptr, y4, HID, NCLS_PAD, NCLS, /*relu=*/0);
        int nz = N_NODES * NCLS;
        zero_f32<<<(nz + TB - 1) / TB, TB, 0, stream>>>(agg4, nz);
        long long st = (long long)N_EDGES * (NCLS / 4);
        scatter_add_f32<<<(unsigned)((st + TB - 1) / TB), TB, 0, stream>>>(y4, ei, agg4, NCLS / 4, NCLS);
        combine_out<<<(nz + TB - 1) / TB, TB, 0, stream>>>(y4, agg4, b4, out);
    }
}
